// LightweightContextMemoryBank_87926570483966
// MI455X (gfx1250) — compile-verified, matrix operand order checked by reading, not run on hardware
//
#include <hip/hip_runtime.h>
#include <cstdint>

#define HW     4096          // 64*64
#define C_IN   640
#define KDQ    160
#define R_MEM  4
#define B_SZ   8
#define K_SEL  2
#define TEMP   0.1f

typedef __attribute__((ext_vector_type(16))) __bf16 v16bf;
typedef __attribute__((ext_vector_type(4)))  __bf16 bf16x4;
typedef __attribute__((ext_vector_type(8)))  float  v8f;

// ---------------- kernel 1: spatial mean -> xm[b*C+c] ----------------
__global__ void mean_kernel(const float* __restrict__ ctx, float* __restrict__ xm) {
  __shared__ float red[256];
  const int tid = threadIdx.x;
  const float4* src = (const float4*)(ctx + (size_t)blockIdx.x * HW);
  float s = 0.f;
  #pragma unroll
  for (int i = 0; i < 4; ++i) {
    float4 v = src[tid + i * 256];
    s += (v.x + v.y) + (v.z + v.w);
  }
  red[tid] = s; __syncthreads();
  for (int off = 128; off > 0; off >>= 1) {
    if (tid < off) red[tid] += red[tid + off];
    __syncthreads();
  }
  if (tid == 0) xm[blockIdx.x] = red[0] * (1.0f / (float)HW);
}

// ---------- kernel 2: query, cosine sims, top-2, softmax(T=0.1) ----------
__global__ void attn_kernel(const float* __restrict__ xm, const float* __restrict__ Wq,
                            const float* __restrict__ bq, const float* __restrict__ keys,
                            float* __restrict__ attn, int* __restrict__ topi) {
  __shared__ float q[KDQ];
  const int b = blockIdx.x, tid = threadIdx.x;
  if (tid < KDQ) {
    float s = 0.f;
    const float* xr = xm + b * C_IN;
    const float* wr = Wq + tid * C_IN;
    for (int c = 0; c < C_IN; ++c) s += xr[c] * wr[c];
    q[tid] = s + bq[tid];
  }
  __syncthreads();
  if (tid == 0) {
    float qn2 = 0.f;
    for (int i = 0; i < KDQ; ++i) qn2 += q[i] * q[i];
    float qn = fmaxf(sqrtf(qn2), 1e-12f);
    float s[R_MEM];
    for (int r = 0; r < R_MEM; ++r) {
      float kk = 0.f, dot = 0.f;
      for (int i = 0; i < KDQ; ++i) {
        float kv = keys[r * KDQ + i];
        kk += kv * kv; dot += kv * q[i];
      }
      s[r] = dot / (qn * fmaxf(sqrtf(kk), 1e-12f));
    }
    int i1 = 0;
    for (int r = 1; r < R_MEM; ++r) if (s[r] > s[i1]) i1 = r;
    int i2 = -1;
    for (int r = 0; r < R_MEM; ++r) {
      if (r == i1) continue;
      if (i2 < 0 || s[r] > s[i2]) i2 = r;
    }
    float e2 = expf((s[i2] - s[i1]) / TEMP);   // stable: s[i1] >= s[i2]
    float inv = 1.f / (1.f + e2);
    attn[b * 2 + 0] = inv;
    attn[b * 2 + 1] = e2 * inv;
    topi[b * 2 + 0] = i1;
    topi[b * 2 + 1] = i2;
  }
}

// ------ kernel 3: gather + bilinear 16->64 upsample + attn-weighted sum ------
__global__ void wref_kernel(const float* __restrict__ vc, const float* __restrict__ attn,
                            const int* __restrict__ topi, float* __restrict__ wref) {
  const size_t id = (size_t)blockIdx.x * 256 + threadIdx.x;   // over B*C*H*W
  const int w = (int)(id & 63);
  const int h = (int)((id >> 6) & 63);
  const int bc = (int)(id >> 12);
  const int b = bc / C_IN, c = bc - b * C_IN;

  const float fy = (h + 0.5f) * 0.25f - 0.5f;      // half-pixel centers
  const float fx = (w + 0.5f) * 0.25f - 0.5f;
  const float y0f = floorf(fy), x0f = floorf(fx);
  const float wy = fy - y0f, wx = fx - x0f;
  const int y0 = min(15, max(0, (int)y0f)), y1 = min(15, max(0, (int)y0f + 1));
  const int x0 = min(15, max(0, (int)x0f)), x1 = min(15, max(0, (int)x0f + 1));

  float acc = 0.f;
  #pragma unroll
  for (int j = 0; j < K_SEL; ++j) {
    const int r = topi[b * 2 + j];
    const float* base = vc + (((size_t)(r * B_SZ + b) * C_IN + c) << 8);
    float v00 = base[y0 * 16 + x0], v01 = base[y0 * 16 + x1];
    float v10 = base[y1 * 16 + x0], v11 = base[y1 * 16 + x1];
    float v = (1.f - wy) * ((1.f - wx) * v00 + wx * v01)
            +        wy  * ((1.f - wx) * v10 + wx * v11);
    acc += attn[b * 2 + j] * v;
  }
  wref[id] = acc;
}

// ---- kernel 4: fused gate GEMM (bf16 WMMA) + sigmoid + blend ----
// D[m,n] = sum_d ctx[m,d]*Wg[n,d] + sum_d wref[m,d]*Wg[n,640+d] + bg[n]
// m=(b,hw) (M=32768), n=channel (N=640), unified K=1280. 128x64 block tile,
// 8 waves x (2x2) 16x16 WMMA tiles, LDS double buffered, transposed f32 epilogue.
#define BM   128
#define BN   64
#define BK   32
#define NSTEP (2 * C_IN / BK)     // 40
#define LDA  40                   // ushort stride, padded
#define ABUF 10240                // 128*40*2 bytes
#define BBUF 5120                 // 64*40*2 bytes
#define LDBUF (ABUF + BBUF)       // 15360 per buffer
#define LDT  132                  // f32 stride of transposed D stage [64][132]

__device__ __forceinline__ unsigned int pkbf(float a, float b) {
  unsigned short ua = __builtin_bit_cast(unsigned short, (__bf16)a);
  unsigned short ub = __builtin_bit_cast(unsigned short, (__bf16)b);
  return (unsigned int)ua | ((unsigned int)ub << 16);
}

__device__ __forceinline__ float gateblend(float pre, float x, float rf) {
  const float g = 1.f / (1.f + expf(-pre));
  return g * rf + (1.f - g) * x;
}

__global__ __launch_bounds__(256)
void gate_fuse_kernel(const float* __restrict__ ctx, const float* __restrict__ wref,
                      const float* __restrict__ Wg, const float* __restrict__ bg,
                      float* __restrict__ out) {
  __shared__ __align__(16) unsigned char smem[33792];   // max(2*15360, 64*132*4)

  const int tid = threadIdx.x;
  const int m0  = blockIdx.x * BM;
  const int n0  = blockIdx.y * BN;
  const int b   = m0 >> 12;            // 4096 % 128 == 0 => tile within one image
  const int hw0 = m0 & 4095;
  const size_t bbase = (size_t)b * C_IN * HW + hw0;

  const int lane = tid & 31;
  const int wave = tid >> 5;
  const int mt = (wave & 3) * 32;      // wave m-tiles: mt, mt+16
  const int nt = (wave >> 2) * 32;     // wave n-tiles: nt, nt+16
  const int r  = lane & 15;
  const int hh = lane >> 4;

  v8f acc00 = {0.f,0.f,0.f,0.f,0.f,0.f,0.f,0.f};
  v8f acc01 = acc00, acc10 = acc00, acc11 = acc00;

  // A loader: 128 hw x 32 d; thread owns 4 hw (float4) x 2 d-pairs
  const int a_hw4 = (tid & 31) * 4;    // 0..124
  const int a_dp  = tid >> 5;          // 0..7 -> pairs a_dp, a_dp+8
  // B loader: 64 n x 32 d; thread owns 4 d (float4) x 2 n values
  const int b_d4  = (tid & 7) * 4;     // 0..28
  const int b_n   = tid >> 3;          // 0..31 -> n, n+32

  auto load_tile = [&](int d0, int buf) {
    unsigned short* sA = (unsigned short*)(smem + buf * LDBUF);
    unsigned short* sB = (unsigned short*)(smem + buf * LDBUF + ABUF);
    const float* src = ((d0 < C_IN) ? ctx : wref)
                     + bbase + (size_t)(d0 - (d0 < C_IN ? 0 : C_IN)) * HW;
    #pragma unroll
    for (int i = 0; i < 2; ++i) {
      const int dl = (a_dp + i * 8) * 2;
      float4 v0 = *(const float4*)(src + (size_t)dl * HW + a_hw4);
      float4 v1 = *(const float4*)(src + (size_t)(dl + 1) * HW + a_hw4);
      *(unsigned int*)(sA + (a_hw4 + 0) * LDA + dl) = pkbf(v0.x, v1.x);
      *(unsigned int*)(sA + (a_hw4 + 1) * LDA + dl) = pkbf(v0.y, v1.y);
      *(unsigned int*)(sA + (a_hw4 + 2) * LDA + dl) = pkbf(v0.z, v1.z);
      *(unsigned int*)(sA + (a_hw4 + 3) * LDA + dl) = pkbf(v0.w, v1.w);
    }
    #pragma unroll
    for (int i = 0; i < 2; ++i) {
      const int nl = b_n + i * 32;
      float4 v = *(const float4*)(&Wg[(size_t)(n0 + nl) * (2 * C_IN) + d0 + b_d4]);
      bf16x4 pk = { (__bf16)v.x, (__bf16)v.y, (__bf16)v.z, (__bf16)v.w };
      *(bf16x4*)(sB + nl * LDA + b_d4) = pk;
    }
    if (d0 + BK < 2 * C_IN)   // gfx1250 global_prefetch_b8 on upcoming weight tile
      __builtin_prefetch(&Wg[(size_t)(n0 + b_n) * (2 * C_IN) + d0 + BK + b_d4], 0, 3);
  };

  load_tile(0, 0);
  __syncthreads();

  for (int step = 0; step < NSTEP; ++step) {
    const int cur = step & 1;
    if (step + 1 < NSTEP) load_tile((step + 1) * BK, cur ^ 1);

    const unsigned short* sA = (const unsigned short*)(smem + cur * LDBUF);
    const unsigned short* sB = (const unsigned short*)(smem + cur * LDBUF + ABUF);

    // Fragments per ISA 7.12.2: lane half holds K blocks {h*8..+7, 16+h*8..+7}
    union { uint4 q[2]; v16bf v; } fa0, fa1, fb0, fb1;
    fa0.q[0] = *(const uint4*)(sA + (mt + r) * LDA + hh * 8);
    fa0.q[1] = *(const uint4*)(sA + (mt + r) * LDA + 16 + hh * 8);
    fa1.q[0] = *(const uint4*)(sA + (mt + 16 + r) * LDA + hh * 8);
    fa1.q[1] = *(const uint4*)(sA + (mt + 16 + r) * LDA + 16 + hh * 8);
    fb0.q[0] = *(const uint4*)(sB + (nt + r) * LDA + hh * 8);
    fb0.q[1] = *(const uint4*)(sB + (nt + r) * LDA + 16 + hh * 8);
    fb1.q[0] = *(const uint4*)(sB + (nt + 16 + r) * LDA + hh * 8);
    fb1.q[1] = *(const uint4*)(sB + (nt + 16 + r) * LDA + 16 + hh * 8);

    acc00 = __builtin_amdgcn_wmma_f32_16x16x32_bf16(false, fa0.v, false, fb0.v,
                                                    (short)0, acc00, false, false);
    acc01 = __builtin_amdgcn_wmma_f32_16x16x32_bf16(false, fa0.v, false, fb1.v,
                                                    (short)0, acc01, false, false);
    acc10 = __builtin_amdgcn_wmma_f32_16x16x32_bf16(false, fa1.v, false, fb0.v,
                                                    (short)0, acc10, false, false);
    acc11 = __builtin_amdgcn_wmma_f32_16x16x32_bf16(false, fa1.v, false, fb1.v,
                                                    (short)0, acc11, false, false);
    __syncthreads();
  }

  // Stage D transposed: sD[n][m], so lane's 8 contiguous m rows -> 2 ds_store_b128
  float* sD = (float*)smem;
  {
    union { v8f v; float4 q[2]; } c00, c01, c10, c11;
    c00.v = acc00; c01.v = acc01; c10.v = acc10; c11.v = acc11;
    const int row0 = mt + hh * 8;
    *(float4*)(sD + (nt + r) * LDT + row0)           = c00.q[0];
    *(float4*)(sD + (nt + r) * LDT + row0 + 4)       = c00.q[1];
    *(float4*)(sD + (nt + 16 + r) * LDT + row0)      = c01.q[0];
    *(float4*)(sD + (nt + 16 + r) * LDT + row0 + 4)  = c01.q[1];
    *(float4*)(sD + (nt + r) * LDT + row0 + 16)      = c10.q[0];
    *(float4*)(sD + (nt + r) * LDT + row0 + 20)      = c10.q[1];
    *(float4*)(sD + (nt + 16 + r) * LDT + row0 + 16) = c11.q[0];
    *(float4*)(sD + (nt + 16 + r) * LDT + row0 + 20) = c11.q[1];
  }
  __syncthreads();

  // Epilogue: one ds_load_b128 of gate_pre + b128 global loads/stores per float4
  const int e_hw4 = (tid & 31) * 4;    // 0..124
  const int e_n   = tid >> 5;          // 0..7 -> nl = e_n + 8*i
  #pragma unroll
  for (int i = 0; i < 8; ++i) {
    const int nl  = e_n + i * 8;
    const int cch = n0 + nl;
    const float bgv = bg[cch];
    float4 pre = *(const float4*)(sD + nl * LDT + e_hw4);
    const size_t addr = (size_t)b * C_IN * HW + (size_t)cch * HW + hw0 + e_hw4;
    float4 x  = *(const float4*)(ctx + addr);
    float4 rf = *(const float4*)(wref + addr);
    float4 o;
    o.x = gateblend(pre.x + bgv, x.x, rf.x);
    o.y = gateblend(pre.y + bgv, x.y, rf.y);
    o.z = gateblend(pre.z + bgv, x.z, rf.z);
    o.w = gateblend(pre.w + bgv, x.w, rf.w);
    *(float4*)(out + addr) = o;
  }
}

extern "C" void kernel_launch(void* const* d_in, const int* in_sizes, int n_in,
                              void* d_out, int out_size, void* d_ws, size_t ws_size,
                              hipStream_t stream) {
  const float* ctx  = (const float*)d_in[0];
  const float* Wq   = (const float*)d_in[1];
  const float* bq   = (const float*)d_in[2];
  const float* keys = (const float*)d_in[3];
  const float* vc   = (const float*)d_in[4];
  const float* Wg   = (const float*)d_in[5];
  const float* bg   = (const float*)d_in[6];
  float* out = (float*)d_out;

  // workspace layout (floats): xm[5120] | attn[16] | topi[16 ints] | pad | wref[B*C*H*W]
  float* ws   = (float*)d_ws;
  float* xm   = ws;
  float* attn = ws + 5120;
  int*   topi = (int*)(ws + 5152);
  float* wref = ws + 8192;                       // ~84 MB, needs ws_size >= ~84.1 MB

  mean_kernel<<<B_SZ * C_IN, 256, 0, stream>>>(ctx, xm);
  attn_kernel<<<B_SZ, 256, 0, stream>>>(xm, Wq, bq, keys, attn, topi);
  wref_kernel<<<(B_SZ * C_IN * HW) / 256, 256, 0, stream>>>(vc, attn, topi, wref);

  dim3 grid((B_SZ * HW) / BM, C_IN / BN);        // 256 x 10
  gate_fuse_kernel<<<grid, 256, 0, stream>>>(ctx, wref, Wg, bg, out);
}
